// ConvCaps3D_40020505264452
// MI455X (gfx1250) — matrix operand order repaired and checked by reading, not compiled, not run on hardware
//
#include <hip/hip_runtime.h>

typedef __attribute__((ext_vector_type(16))) __bf16 v16bf;
typedef __attribute__((ext_vector_type(8)))  float  v8f;

#define B_N    8
#define C_IN   16
#define D_CAP  32     // input capsule depth (N of the GEMM)
#define H_IN   24
#define W_IN   24
#define C_OUTC 16
#define D_OUTC 32
#define KK     3
#define H_OUT  22
#define W_OUT  22
#define M_DIM  512    // C_OUTC * D_OUTC
#define K_DIM  144    // C_IN * 3 * 3
#define K_PAD  160    // 5 tiles of 32
#define ROUT_ITERS 3

#define WF_ELEMS (32*5*32*16)                       // 81920 bf16
#define XT_ELEMS (B_N*H_IN*W_IN*C_IN*D_CAP)        // 2359296 bf16
#define WF_BYTES (WF_ELEMS*2)                       // 163840, 256B-aligned

// ---------------------------------------------------------------------------
// Pre-swizzle weights into WMMA A-fragment layout (bf16), K padded to 160.
// wf[((mt*5+kt)*32+lane)*16+j]; contiguous K order (same as B frag):
//   lane<16 : K=j
//   lane>=16: K=j+16
// ---------------------------------------------------------------------------
__global__ void caps_pack_weights(const float* __restrict__ w, __bf16* __restrict__ wf) {
  int idx = blockIdx.x * 256 + threadIdx.x;
  if (idx >= WF_ELEMS) return;
  int j    = idx & 15;
  int lane = (idx >> 4) & 31;
  int kt   = (idx >> 9) % 5;
  int mt   = idx / 2560;
  int m = mt * 16 + (lane & 15);
  int kin = (lane < 16) ? j : (j + 16);
  int k = kt * 32 + kin;
  float v = (k < K_DIM) ? w[m * K_DIM + k] : 0.0f;
  wf[idx] = (__bf16)v;
}

// ---------------------------------------------------------------------------
// Transpose x (B,C,D,H,W) fp32 -> xt (B,y,x,Cin,D) bf16 so D is contiguous.
// ---------------------------------------------------------------------------
__global__ void caps_pack_x(const float* __restrict__ x, __bf16* __restrict__ xt) {
  int idx = blockIdx.x * 256 + threadIdx.x;
  if (idx >= XT_ELEMS) return;
  int Dd  = idx & 31;
  int cin = (idx >> 5) & 15;
  int t   = idx >> 9;
  int xw  = t % W_IN; t /= W_IN;
  int y   = t % H_IN;
  int b   = t / H_IN;
  float v = x[(((b * C_IN + cin) * D_CAP + Dd) * H_IN + y) * W_IN + xw];
  xt[idx] = (__bf16)v;
}

// ---------------------------------------------------------------------------
// Fused conv-GEMM (WMMA bf16) + dynamic routing, one block per (B,h,w).
// ---------------------------------------------------------------------------
__global__ __launch_bounds__(256) void caps_fused_kernel(
    const __bf16* __restrict__ wf, const __bf16* __restrict__ xt,
    const float* __restrict__ bias, const float* __restrict__ prior,
    float* __restrict__ out) {

  __shared__ __align__(32) __bf16 sU[M_DIM * D_CAP];   // u[m][D], 32 KB
  __shared__ __align__(32) __bf16 sB[D_CAP * K_PAD];   // patch [D][k], 10 KB
  __shared__ float sbb[D_OUTC * D_CAP];                // routing logits b
  __shared__ float scc[D_OUTC * D_CAP];                // softmax cc
  __shared__ float ssv[C_OUTC * D_OUTC];               // s / v
  __shared__ float sn2[D_OUTC];                        // squash scale

  const int t    = threadIdx.x;
  const int lane = t & 31;
  const int wv   = t >> 5;

  const int pos = blockIdx.x;
  const int b   = pos / (H_OUT * W_OUT);
  const int rem = pos % (H_OUT * W_OUT);
  const int h   = rem / W_OUT;
  const int w   = rem % W_OUT;

  // ---- build patch B-matrix in LDS: sB[Dd*K_PAD + k] (coalesced over D) ----
  for (int idx = t; idx < D_CAP * K_PAD; idx += 256) {
    int Dd = idx & 31;
    int k  = idx >> 5;
    __bf16 v = (__bf16)0.0f;
    if (k < K_DIM) {
      int cin = k / 9;
      int r   = k - cin * 9;
      int kh  = r / 3, kw = r - kh * 3;
      v = xt[((b * H_IN + (h + kh)) * W_IN + (w + kw)) * (C_IN * D_CAP)
             + cin * D_CAP + Dd];
    }
    sB[Dd * K_PAD + k] = v;
  }
  // ---- init routing logits from prior while we're here ----
  for (int idx = t; idx < D_OUTC * D_CAP; idx += 256) sbb[idx] = prior[idx];
  __syncthreads();

  // ---- GEMM: wave wv owns M-tiles 4wv..4wv+3, both N-tiles ----
  v8f acc[4][2];
  {
    v8f z = {};
    for (int i = 0; i < 4; i++) { acc[i][0] = z; acc[i][1] = z; }
  }
  const int nrow  = lane & 15;
  const int khalf = (lane < 16) ? 0 : 16;   // B frag: lane<16 K 0..15, else 16..31
  #pragma unroll
  for (int kt = 0; kt < 5; kt++) {
    v16bf bf0 = *(const v16bf*)&sB[(0 * 16 + nrow) * K_PAD + kt * 32 + khalf];
    v16bf bf1 = *(const v16bf*)&sB[(1 * 16 + nrow) * K_PAD + kt * 32 + khalf];
    #pragma unroll
    for (int i = 0; i < 4; i++) {
      int mt = 4 * wv + i;
      v16bf af = *(const v16bf*)&wf[((mt * 5 + kt) * 32 + lane) * 16];
      acc[i][0] = __builtin_amdgcn_wmma_f32_16x16x32_bf16(
          false, af, false, bf0, (short)0, acc[i][0], false, false);
      acc[i][1] = __builtin_amdgcn_wmma_f32_16x16x32_bf16(
          false, af, false, bf1, (short)0, acc[i][1], false, false);
    }
  }

  // ---- write u (+bias) to LDS as bf16; C layout: VGPR v -> M=v (lanes<16) / v+8 ----
  const int mrow_base = (lane < 16) ? 0 : 8;
  #pragma unroll
  for (int i = 0; i < 4; i++) {
    int mt = 4 * wv + i;
    #pragma unroll
    for (int n = 0; n < 2; n++) {
      int Dd = n * 16 + nrow;
      #pragma unroll
      for (int v = 0; v < 8; v++) {
        int m = mt * 16 + mrow_base + v;
        float val = acc[i][n][v] + bias[m];
        sU[m * D_CAP + Dd] = (__bf16)val;
      }
    }
  }
  __syncthreads();

  // ---- dynamic routing (3 iterations) entirely in LDS ----
  for (int it = 0; it < ROUT_ITERS; it++) {
    // softmax over d for each column D (32 columns, one thread each)
    if (t < D_CAP) {
      int Dd = t;
      float mx = -1e30f;
      for (int d = 0; d < D_OUTC; d++) mx = fmaxf(mx, sbb[d * D_CAP + Dd]);
      float sum = 0.f;
      for (int d = 0; d < D_OUTC; d++) sum += expf(sbb[d * D_CAP + Dd] - mx);
      float inv = 1.0f / sum;
      for (int d = 0; d < D_OUTC; d++)
        scc[d * D_CAP + Dd] = expf(sbb[d * D_CAP + Dd] - mx) * inv;
    }
    __syncthreads();
    // s[c][d] = sum_D cc[d][D] * u[c][d][D]   (u rows contiguous)
    for (int idx = t; idx < C_OUTC * D_OUTC; idx += 256) {
      int c = idx >> 5, d = idx & 31;
      const __bf16* up = &sU[(c * D_OUTC + d) * D_CAP];
      const float*  cp = &scc[d * D_CAP];
      float s = 0.f;
      for (int Dd = 0; Dd < D_CAP; Dd++) s += cp[Dd] * (float)up[Dd];
      ssv[idx] = s;
    }
    __syncthreads();
    // squash scale per d
    if (t < D_OUTC) {
      int d = t;
      float n2 = 0.f;
      for (int c = 0; c < C_OUTC; c++) { float sv = ssv[c * D_OUTC + d]; n2 += sv * sv; }
      sn2[d] = (n2 / (1.0f + n2)) * rsqrtf(n2 + 1e-8f);
    }
    __syncthreads();
    for (int idx = t; idx < C_OUTC * D_OUTC; idx += 256) ssv[idx] *= sn2[idx & 31];
    __syncthreads();
    // b += a, a[d][D] = sum_c u[c][d][D] * v[c][d]
    for (int idx = t; idx < D_OUTC * D_CAP; idx += 256) {
      int d = idx >> 5, Dd = idx & 31;
      float a = 0.f;
      for (int c = 0; c < C_OUTC; c++)
        a += (float)sU[(c * D_OUTC + d) * D_CAP + Dd] * ssv[c * D_OUTC + d];
      sbb[idx] += a;
    }
    __syncthreads();
  }

  // ---- final softmax + output einsum straight to global ----
  if (t < D_CAP) {
    int Dd = t;
    float mx = -1e30f;
    for (int d = 0; d < D_OUTC; d++) mx = fmaxf(mx, sbb[d * D_CAP + Dd]);
    float sum = 0.f;
    for (int d = 0; d < D_OUTC; d++) sum += expf(sbb[d * D_CAP + Dd] - mx);
    float inv = 1.0f / sum;
    for (int d = 0; d < D_OUTC; d++)
      scc[d * D_CAP + Dd] = expf(sbb[d * D_CAP + Dd] - mx) * inv;
  }
  __syncthreads();
  for (int idx = t; idx < C_OUTC * D_OUTC; idx += 256) {
    int c = idx >> 5, d = idx & 31;
    const __bf16* up = &sU[(c * D_OUTC + d) * D_CAP];
    const float*  cp = &scc[d * D_CAP];
    float s = 0.f;
    for (int Dd = 0; Dd < D_CAP; Dd++) s += cp[Dd] * (float)up[Dd];
    out[((b * C_OUTC + c) * D_OUTC + d) * (H_OUT * W_OUT) + h * W_OUT + w] = s;
  }
}

extern "C" void kernel_launch(void* const* d_in, const int* in_sizes, int n_in,
                              void* d_out, int out_size, void* d_ws, size_t ws_size,
                              hipStream_t stream) {
  const float* x     = (const float*)d_in[0];   // (8,16,32,24,24)
  const float* wk    = (const float*)d_in[1];   // (512,16,1,3,3)
  const float* wb    = (const float*)d_in[2];   // (512,)
  const float* prior = (const float*)d_in[3];   // (1,32,32,1,1) -> 1024
  float* out = (float*)d_out;

  __bf16* wf = (__bf16*)d_ws;                            // 160 KB fragment-packed W
  __bf16* xt = (__bf16*)((char*)d_ws + WF_BYTES);        // 4.5 MB transposed x

  caps_pack_weights<<<(WF_ELEMS + 255) / 256, 256, 0, stream>>>(wk, wf);
  caps_pack_x<<<(XT_ELEMS + 255) / 256, 256, 0, stream>>>(x, xt);
  caps_fused_kernel<<<B_N * H_OUT * W_OUT, 256, 0, stream>>>(wf, xt, wb, prior, out);
}
